// ModuleSepconv_8641474199683
// MI455X (gfx1250) — matrix operand order repaired and checked by reading, not compiled, not run on hardware
//
#include <hip/hip_runtime.h>
#include <stdint.h>

typedef __attribute__((ext_vector_type(16))) _Float16 v16h;
typedef __attribute__((ext_vector_type(8)))  float    v8f;

#define BB    2
#define CC    3
#define HH    256
#define WW    256
#define KK    51
#define WIN   306     // W + K - 1
#define TX    16      // output pixels per tile (GEMM M)
#define TY    8       // output rows per workgroup
#define UC    66      // patch columns = TX + KK - 1
#define PROWS 60      // padded patch rows (58 valid: KK-1+TY)
#define NTILE 5       // ceil(UC/16)
#define NTHR  96      // 3 waves: one per channel

// dynamic LDS layout (all f32 staging; converted to f16 at fragment build)
#define OFF_P 0                                  // [CC][UC][PROWS] f32  = 47520 B
#define OFF_V (OFF_P + CC*UC*PROWS*4)            // [TY][TX][52]   f32  = 26624 B
#define OFF_H (OFF_V + TY*TX*52*4)               // [TY][TX][52]   f32  = 26624 B
#define OFF_Q (OFF_H + TY*TX*52*4)               // [CC][TX][80]   f32  = 15360 B
#define SMEM_BYTES (OFF_Q + CC*TX*80*4)          // 116128 B dynamic LDS

__device__ __forceinline__ void async_ld_b32(uint32_t lds_byte_addr,
                                             uint32_t glb_byte_off,
                                             const float* sbase) {
    // GLOBAL_LOAD_ASYNC_TO_LDS_B32 (GVS mode): LDS[vdst] = MEM[saddr + vaddr]
    asm volatile("global_load_async_to_lds_b32 %0, %1, %2"
                 :: "v"(lds_byte_addr), "v"(glb_byte_off),
                    "s"((uint64_t)(uintptr_t)sbase)
                 : "memory");
}

__global__ __launch_bounds__(NTHR)
void sepconv_wmma_async(const float* __restrict__ inp,
                        const float* __restrict__ vert,
                        const float* __restrict__ horz,
                        float* __restrict__ out) {
    extern __shared__ char smem[];
    float* Pf = (float*)(smem + OFF_P);
    float* Vf = (float*)(smem + OFF_V);
    float* Hf = (float*)(smem + OFF_H);
    float* Qf = (float*)(smem + OFF_Q);
    const uint32_t lds0 = (uint32_t)(uintptr_t)smem;   // LDS byte offset of smem

    const int tid = threadIdx.x;
    const int bb  = blockIdx.z;
    const int y0  = blockIdx.y * TY;
    const int x0  = blockIdx.x * TX;

    // ---------- async patch staging: rows y0..y0+57, cols x0..x0+65, 3 ch ----------
    {
        int u0 = tid, r0 = 0;
        if (u0 >= UC) { u0 -= UC; r0 = 1; }
        const uint32_t vo0 = (uint32_t)(r0 * WIN + u0) * 4u;
        for (int c = 0; c < CC; ++c) {
            const float* base = inp + ((size_t)(bb * CC + c) * WIN + y0) * WIN + x0;
            int u = u0, r = r0;
            uint32_t vo = vo0;
            while (r < 58) {
                uint32_t lds = lds0 + OFF_P +
                               (uint32_t)(((c * UC + u) * PROWS + r) * 4);
                async_ld_b32(lds, vo, base);
                // advance flat index by 96 over a width-66 plane: (+1 row, +30 col)
                r += 1; u += 30; vo += (WIN + 30) * 4;
                if (u >= UC) { u -= UC; r += 1; vo += (WIN - UC) * 4; }
            }
        }
    }
    // ---------- async V/H staging: [yi][m][k] ----------
    {
        const int m  = tid & 15;
        const int k0 = tid >> 4;                  // 0..5, stride 6 over k
        for (int yi = 0; yi < TY; ++yi) {
            const float* vbase = vert + ((size_t)(bb * KK) * HH + (y0 + yi)) * WW + x0;
            const float* hbase = horz + ((size_t)(bb * KK) * HH + (y0 + yi)) * WW + x0;
            uint32_t vo   = (uint32_t)((k0 * HH * WW + m) * 4);
            uint32_t ldsV = lds0 + OFF_V + (uint32_t)(((yi * TX + m) * 52 + k0) * 4);
            uint32_t ldsH = lds0 + OFF_H + (uint32_t)(((yi * TX + m) * 52 + k0) * 4);
            for (int k = k0; k < KK; k += 6) {
                async_ld_b32(ldsV, vo, vbase);
                async_ld_b32(ldsH, vo, hbase);
                vo += 6u * HH * WW * 4u;
                ldsV += 6 * 4; ldsH += 6 * 4;
            }
        }
    }
    asm volatile("s_wait_asynccnt 0x0" ::: "memory");
    __syncthreads();

    // ---------- per-wave compute: wave index == channel ----------
    const int wave = tid >> 5;
    const int lane = tid & 31;
    const int m    = lane & 15;        // A row / B col / C col
    const int g    = lane >> 4;        // lane half-group
    float* Pc = Pf + wave * UC * PROWS;
    float* Qc = Qf + wave * TX * 80;

    for (int yi = 0; yi < TY; ++yi) {
        v8f acc0 = {}, acc1 = {}, acc2 = {}, acc3 = {}, acc4 = {};
        v8f* accs[NTILE] = { &acc0, &acc1, &acc2, &acc3, &acc4 };

        #pragma unroll
        for (int s = 0; s < 2; ++s) {
            const int kbase = 32 * s;
            // A fragment (16x32 f16): K = 16g + e
            v16h a;
            #pragma unroll
            for (int e = 0; e < 16; ++e) {
                int kk = kbase + 16 * g + e;
                a[e] = (kk < KK) ? (_Float16)Vf[(yi * TX + m) * 52 + kk]
                                 : (_Float16)0.0f;
            }
            #pragma unroll
            for (int nt = 0; nt < NTILE; ++nt) {
                // B fragment (32x16 f16): lane holds col n=m; K = 16g + e
                int u  = 16 * nt + m;
                int uc = (u < UC) ? u : (UC - 1);       // padded cols: unread Q
                v16h bfrag;
                #pragma unroll
                for (int e = 0; e < 16; ++e) {
                    int row = yi + kbase + 16 * g + e;  // tap = kbase+16g+e
                    row = (row < 57) ? row : 57;        // clamp to last WRITTEN row
                    bfrag[e] = (_Float16)Pc[uc * PROWS + row];
                }
                *accs[nt] = __builtin_amdgcn_wmma_f32_16x16x32_f16(
                    false, a, false, bfrag, (short)0, *accs[nt], false, false);
            }
        }

        // spill Q (C/D layout: VGPR r -> row r+8g, lane -> col 16*nt + m)
        #pragma unroll
        for (int nt = 0; nt < NTILE; ++nt) {
            v8f q = *accs[nt];
            #pragma unroll
            for (int r = 0; r < 8; ++r)
                Qc[(r + 8 * g) * 80 + 16 * nt + m] = q[r];
        }
        __syncthreads();

        // horizontal stage: out[m] = sum_j H[m][j] * Q[m][m+j]  (f32 taps)
        if (lane < 16) {
            float o = 0.0f;
            #pragma unroll 3
            for (int j = 0; j < KK; ++j)
                o += Hf[(yi * TX + m) * 52 + j] * Qc[m * 80 + m + j];
            out[((size_t)(bb * CC + wave) * HH + (y0 + yi)) * WW + x0 + m] = o;
        }
        __syncthreads();   // Q reused next yi
    }
}

extern "C" void kernel_launch(void* const* d_in, const int* in_sizes, int n_in,
                              void* d_out, int out_size, void* d_ws, size_t ws_size,
                              hipStream_t stream) {
    (void)in_sizes; (void)n_in; (void)d_ws; (void)ws_size; (void)out_size;
    const float* inp  = (const float*)d_in[0];
    const float* vert = (const float*)d_in[1];
    const float* horz = (const float*)d_in[2];
    float* out = (float*)d_out;

    dim3 grid(WW / TX, HH / TY, BB);   // 16 x 32 x 2
    dim3 block(NTHR);                  // 3 waves: one per channel
    sepconv_wmma_async<<<grid, block, SMEM_BYTES, stream>>>(inp, vert, horz, out);
}
